// HyperGNNLayer_10290741641950
// MI455X (gfx1250) — compile-verified, matrix operand order checked by reading, not run on hardware
//
#include <hip/hip_runtime.h>
#include <hip/hip_bf16.h>

// ---------------- problem constants ----------------
#define BQ       16
#define N_NODES  2304
#define NROWS    (BQ * N_NODES)     // 36864
#define H_DIM    256
#define E_EDGES  131072
#define TM       32                  // rows per workgroup tile
#define LDSS     264                 // LDS row stride in halfs (16B-aligned, padded)

typedef _Float16 half16 __attribute__((ext_vector_type(16)));
typedef float    float8 __attribute__((ext_vector_type(8)));

// Load a 16-half WMMA fragment: 8 halfs at p[0..7] and 8 at p[16..23]
// (CDNA5 16-bit A/B per-lane layout: K = kb..kb+7 and kb+16..kb+23).
__device__ __forceinline__ half16 load_frag(const _Float16* p) {
    union { float4 f[2]; half16 h; } u;
    u.f[0] = *(const float4*)(p);
    u.f[1] = *(const float4*)(p + 16);
    return u.h;
}

// One 32x256 @ 256x256 GEMM slice for this wave: 2 M-tiles x 2 N-tiles
// (each B fragment fetched from L2 feeds two WMMAs -> 2x reuse).
// sIn: LDS activations (TM x LDSS halfs). Wt: f16 weights, transposed (N x K row-major).
__device__ __forceinline__ void mm_tiles(const _Float16* sIn, const _Float16* Wt,
                                         int lane, int nw, float8 acc[2][2]) {
    const int ar = lane & 15;
    const int kb = (lane >> 4) * 8;
#pragma unroll
    for (int mi = 0; mi < 2; ++mi)
#pragma unroll
        for (int j = 0; j < 2; ++j) acc[mi][j] = (float8)(0.0f);
#pragma unroll
    for (int k = 0; k < 8; ++k) {
        half16 a0 = load_frag(sIn + ar * LDSS + k * 32 + kb);
        half16 a1 = load_frag(sIn + (16 + ar) * LDSS + k * 32 + kb);
#pragma unroll
        for (int j = 0; j < 2; ++j) {
            const int n = (nw * 2 + j) * 16 + ar;
            half16 b = load_frag(Wt + n * 256 + k * 32 + kb);
            acc[0][j] = __builtin_amdgcn_wmma_f32_16x16x32_f16(
                false, a0, false, b, (short)0, acc[0][j], false, false);
            acc[1][j] = __builtin_amdgcn_wmma_f32_16x16x32_f16(
                false, a1, false, b, (short)0, acc[1][j], false, false);
        }
    }
}

// Bias + (optional) ReLU, write f16 tile back to LDS in C/D layout.
__device__ __forceinline__ void store_act(_Float16* sOut, const float* bias,
                                          int lane, int nw,
                                          const float8 acc[2][2], bool relu) {
    const int rb = 8 * (lane >> 4);
    const int cl = lane & 15;
#pragma unroll
    for (int mi = 0; mi < 2; ++mi) {
        const int rbase = mi * 16 + rb;
#pragma unroll
        for (int j = 0; j < 2; ++j) {
            const int c  = (nw * 2 + j) * 16 + cl;
            const float bv = bias[c];
#pragma unroll
            for (int r = 0; r < 8; ++r) {
                float v = acc[mi][j][r] + bv;
                if (relu) v = fmaxf(v, 0.0f);
                sOut[(rbase + r) * LDSS + c] = (_Float16)v;
            }
        }
    }
}

// ---------------- fused LN + 3-layer MLP + residual / scatter ----------------
// mode 0: Y = X + mlp(LN(X)+pe);                     write Yout[row]
// mode 1: (k1 nodes) Y = (X + mlp)(mask);           write Yout (=x1acc), atomic x0acc
// mode 2: (edges) X = v*Win+bin; Y = X + mlp;       atomic x0acc (*s0), x1acc (*s1)
// mode 3: X = x1[row] + x0[batch]; Y = X + mlp;     Y = Y*mask + xk1; write Yout
__global__ void __launch_bounds__(256)
fused_ln_mlp_kernel(int mode, int M,
                    const float* __restrict__ Xin,
                    const float* __restrict__ vvals,
                    const float* __restrict__ Win, const float* __restrict__ bin,
                    const int* __restrict__ eb, const int* __restrict__ en,
                    const float* __restrict__ lng, const float* __restrict__ lnb,
                    const float* __restrict__ pe,
                    const _Float16* __restrict__ Wt0, const float* __restrict__ b0,
                    const _Float16* __restrict__ Wt1, const float* __restrict__ b1,
                    const _Float16* __restrict__ Wt2, const float* __restrict__ b2,
                    float* __restrict__ Yout, float* __restrict__ x0acc,
                    float s0, float s1,
                    const int* __restrict__ n1, const int* __restrict__ n2,
                    const float* __restrict__ x0row, const float* __restrict__ xk1) {
    __shared__ __align__(16) _Float16 sA[TM * LDSS];
    __shared__ __align__(16) _Float16 sB[TM * LDSS];
    const int tid  = threadIdx.x;
    const int lane = tid & 31;
    const int wid  = tid >> 5;
    const int row0 = blockIdx.x * TM;

    // Pull the later layers' weights toward this WGP while layer 0 runs.
    __builtin_prefetch(Wt1 + (size_t)tid * 256, 0, 0);
    __builtin_prefetch(Wt2 + (size_t)tid * 256, 0, 0);

    // ---- fetch + LayerNorm + pe -> sA (f16) ----
    {
        const int r    = tid >> 3;     // 0..31 : row in tile
        const int sub  = tid & 7;      // 8 threads per row
        const int grow = row0 + r;
        const int c0   = sub * 32;
        float xv[32];
        float s = 0.0f, ss = 0.0f;
        if (grow < M) {
            if (mode == 2) {
                const float v = vvals[grow];
#pragma unroll
                for (int i = 0; i < 32; ++i) {
                    float t = v * Win[c0 + i] + bin[c0 + i];
                    xv[i] = t; s += t; ss += t * t;
                }
            } else if (mode == 3) {
                const int b = grow / N_NODES;
#pragma unroll
                for (int i = 0; i < 32; ++i) {
                    float t = Xin[(size_t)grow * H_DIM + c0 + i] + x0row[b * H_DIM + c0 + i];
                    xv[i] = t; s += t; ss += t * t;
                }
            } else {
#pragma unroll
                for (int i = 0; i < 32; ++i) {
                    float t = Xin[(size_t)grow * H_DIM + c0 + i];
                    xv[i] = t; s += t; ss += t * t;
                }
            }
        } else {
#pragma unroll
            for (int i = 0; i < 32; ++i) xv[i] = 0.0f;
        }
#pragma unroll
        for (int m = 1; m < 8; m <<= 1) {           // reduce across 8 lanes of this row
            s  += __shfl_xor(s, m, 32);
            ss += __shfl_xor(ss, m, 32);
        }
        const float mean = s * (1.0f / 256.0f);
        const float var  = ss * (1.0f / 256.0f) - mean * mean;
        const float rs   = rsqrtf(var + 1e-5f);
#pragma unroll
        for (int i = 0; i < 32; ++i) {
            const int c = c0 + i;
            float t = (xv[i] - mean) * rs * lng[c] + lnb[c];
            if (pe) t += pe[c];
            sA[r * LDSS + c] = (_Float16)t;
        }
    }
    __syncthreads();

    const int nw = wid;           // 8 waves x 2 n-tiles = all 16 n-tiles
    float8 acc[2][2];

    mm_tiles(sA, Wt0, lane, nw, acc);
    store_act(sB, b0, lane, nw, acc, true);
    __syncthreads();
    mm_tiles(sB, Wt1, lane, nw, acc);
    store_act(sA, b1, lane, nw, acc, true);
    __syncthreads();
    mm_tiles(sA, Wt2, lane, nw, acc);

    // ---- epilogue ----
    const int rb = 8 * (lane >> 4);
    const int cl = lane & 15;
#pragma unroll
    for (int mi = 0; mi < 2; ++mi) {
#pragma unroll
        for (int j = 0; j < 2; ++j) {
            const int c  = (nw * 2 + j) * 16 + cl;
            const float bv = b2[c];
#pragma unroll
            for (int r = 0; r < 8; ++r) {
                const int grow = row0 + mi * 16 + rb + r;
                if (grow >= M) continue;
                float y = acc[mi][j][r] + bv;
                if (mode == 0) {
                    y += Xin[(size_t)grow * H_DIM + c];
                    Yout[(size_t)grow * H_DIM + c] = y;
                } else if (mode == 1) {
                    y += Xin[(size_t)grow * H_DIM + c];
                    const int b = grow / N_NODES, node = grow - b * N_NODES;
                    const bool keep = node < n1[b] * n2[b];
                    const float ym = keep ? y : 0.0f;
                    Yout[(size_t)grow * H_DIM + c] = ym;
                    if (keep) atomicAdd(&x0acc[b * H_DIM + c], ym);
                } else if (mode == 2) {
                    y += vvals[grow] * Win[c] + bin[c];
                    const int b = eb[grow], nd = en[grow];
                    atomicAdd(&x0acc[b * H_DIM + c], s0 * y);
                    atomicAdd(&Yout[((size_t)b * N_NODES + nd) * H_DIM + c], s1 * y);
                } else {  // mode 3
                    const int b = grow / N_NODES, node = grow - b * N_NODES;
                    y += Xin[(size_t)grow * H_DIM + c] + x0row[b * H_DIM + c];
                    if (node >= n1[b] * n2[b]) y = 0.0f;
                    y += xk1[(size_t)grow * H_DIM + c];
                    Yout[(size_t)grow * H_DIM + c] = y;
                }
            }
        }
    }
}

// ---------------- plain GEMM: Out = A(Mx256,f32) @ W(256x256) + bias ----------------
__global__ void __launch_bounds__(256)
gemm256_kernel(int M, const float* __restrict__ A, const _Float16* __restrict__ Wt,
               const float* __restrict__ bias, float* __restrict__ Out) {
    __shared__ __align__(16) _Float16 sA[TM * LDSS];
    const int tid  = threadIdx.x;
    const int lane = tid & 31;
    const int wid  = tid >> 5;
    const int row0 = blockIdx.x * TM;
    {
        const int r = tid >> 3, sub = tid & 7, grow = row0 + r, c0 = sub * 32;
#pragma unroll
        for (int i = 0; i < 32; ++i) {
            float t = (grow < M) ? A[(size_t)grow * H_DIM + c0 + i] : 0.0f;
            sA[r * LDSS + c0 + i] = (_Float16)t;
        }
    }
    __syncthreads();
    const int nw = wid;
    float8 acc[2][2];
    mm_tiles(sA, Wt, lane, nw, acc);
    const int rb = 8 * (lane >> 4);
    const int cl = lane & 15;
#pragma unroll
    for (int mi = 0; mi < 2; ++mi) {
#pragma unroll
        for (int j = 0; j < 2; ++j) {
            const int c = (nw * 2 + j) * 16 + cl;
            const float bv = bias[c];
#pragma unroll
            for (int r = 0; r < 8; ++r) {
                const int grow = row0 + mi * 16 + rb + r;
                if (grow < M) Out[(size_t)grow * H_DIM + c] = acc[mi][j][r] + bv;
            }
        }
    }
}

// ---------------- small helper kernels ----------------
__global__ void wconv_kernel(const float* __restrict__ W, _Float16* __restrict__ Wt) {
    const int n = blockIdx.x, k = threadIdx.x;       // Wt[n][k] = W[k][n]
    Wt[n * 256 + k] = (_Float16)W[k * 256 + n];
}

__global__ void edge_count_kernel(int E, const int* __restrict__ eb, const int* __restrict__ en,
                                  float* __restrict__ c0, float* __restrict__ c1) {
    const int e = blockIdx.x * 256 + threadIdx.x;
    if (e < E) {
        atomicAdd(&c0[eb[e]], 1.0f);
        atomicAdd(&c1[eb[e] * N_NODES + en[e]], 1.0f);
    }
}

__global__ void x0_fin_kernel(float* __restrict__ x0, const float* __restrict__ c0k2,
                              const float* __restrict__ c0k3,
                              const int* __restrict__ n1, const int* __restrict__ n2) {
    const int i = blockIdx.x * 256 + threadIdx.x;    // B*H
    const int b = i >> 8;
    const float denom = (float)(n1[b] * n2[b]) + c0k2[b] + c0k3[b];
    x0[i] = x0[i] / denom;
}

__global__ void x1_fin_kernel(float* __restrict__ x1, const float* __restrict__ c1k2,
                              const float* __restrict__ c1k3) {
    const size_t i = (size_t)blockIdx.x * 256 + threadIdx.x;  // B*N*H
    const int row = (int)(i >> 8);
    x1[i] = x1[i] / (1.0f + c1k2[row] + c1k3[row]);
}

// ---------------- orchestration ----------------
#define WTS_BYTES (11u * 65536u * 2u)          // 11 f16 256x256 transposed matrices
#define ACC_FLOATS (4096 + 16 + 16 + NROWS + NROWS)
#define ACC_BYTES_PAD 311552u                  // ACC_FLOATS*4 = 311424, padded to 256B

extern "C" void kernel_launch(void* const* d_in, const int* in_sizes, int n_in,
                              void* d_out, int out_size, void* d_ws, size_t ws_size,
                              hipStream_t stream) {
    const float* x    = (const float*)d_in[0];
    const float* w2v  = (const float*)d_in[1];
    const float* w3v  = (const float*)d_in[2];
    const float* Wx   = (const float*)d_in[3];
    const float* bx   = (const float*)d_in[4];
    const float* W2w  = (const float*)d_in[5];
    const float* b2w  = (const float*)d_in[6];
    const float* W3w  = (const float*)d_in[7];
    const float* b3w  = (const float*)d_in[8];
    const float* m1W  = (const float*)d_in[9];
    const float* m1b  = (const float*)d_in[10];
    const float* m2W  = (const float*)d_in[11];
    const float* m2b  = (const float*)d_in[12];
    const float* m3W  = (const float*)d_in[13];
    const float* m3b  = (const float*)d_in[14];
    const float* pe1  = (const float*)d_in[15];
    const float* pe2  = (const float*)d_in[16];
    const float* Wo   = (const float*)d_in[17];
    const float* bo   = (const float*)d_in[18];
    const float* ln1g = (const float*)d_in[19];
    const float* ln1b = (const float*)d_in[20];
    const float* ln2g = (const float*)d_in[21];
    const float* ln2b = (const float*)d_in[22];
    const float* ln3g = (const float*)d_in[23];
    const float* ln3b = (const float*)d_in[24];
    const int*   w2i  = (const int*)d_in[25];
    const int*   w3i  = (const int*)d_in[26];
    const int*   n1   = (const int*)d_in[27];
    const int*   n2   = (const int*)d_in[28];

    char* w = (char*)d_ws;
    _Float16* wts = (_Float16*)w;
    float* x0acc = (float*)(w + WTS_BYTES);
    float* c0k2  = x0acc + 4096;
    float* c0k3  = c0k2 + 16;
    float* c1k2  = c0k3 + 16;
    float* c1k3  = c1k2 + NROWS;
    float* xk1   = (float*)(w + WTS_BYTES + ACC_BYTES_PAD);
    float* x1acc = xk1 + (size_t)NROWS * H_DIM;
    float* xx    = x1acc + (size_t)NROWS * H_DIM;
    float* out   = (float*)d_out;

    _Float16* wtsX  = wts + 0u * 65536u;
    _Float16* wts1a = wts + 1u * 65536u;   // mlp1 layers
    _Float16* wts1b = wts + 2u * 65536u;
    _Float16* wts1c = wts + 3u * 65536u;
    _Float16* wts2a = wts + 4u * 65536u;   // mlp2
    _Float16* wts2b = wts + 5u * 65536u;
    _Float16* wts2c = wts + 6u * 65536u;
    _Float16* wts3a = wts + 7u * 65536u;   // mlp3
    _Float16* wts3b = wts + 8u * 65536u;
    _Float16* wts3c = wts + 9u * 65536u;
    _Float16* wtsO  = wts + 10u * 65536u;

    // 1) weight conversion (fp32 -> transposed f16)
    wconv_kernel<<<256, 256, 0, stream>>>(Wx, wtsX);
    wconv_kernel<<<256, 256, 0, stream>>>(m1W + 0 * 65536, wts1a);
    wconv_kernel<<<256, 256, 0, stream>>>(m1W + 1 * 65536, wts1b);
    wconv_kernel<<<256, 256, 0, stream>>>(m1W + 2 * 65536, wts1c);
    wconv_kernel<<<256, 256, 0, stream>>>(m2W + 0 * 65536, wts2a);
    wconv_kernel<<<256, 256, 0, stream>>>(m2W + 1 * 65536, wts2b);
    wconv_kernel<<<256, 256, 0, stream>>>(m2W + 2 * 65536, wts2c);
    wconv_kernel<<<256, 256, 0, stream>>>(m3W + 0 * 65536, wts3a);
    wconv_kernel<<<256, 256, 0, stream>>>(m3W + 1 * 65536, wts3b);
    wconv_kernel<<<256, 256, 0, stream>>>(m3W + 2 * 65536, wts3c);
    wconv_kernel<<<256, 256, 0, stream>>>(Wo, wtsO);

    // 2) zero accumulators (x0acc + counts)
    hipMemsetAsync(x0acc, 0, (size_t)ACC_FLOATS * sizeof(float), stream);

    // 3) x_k1 = x @ Wx + bx
    gemm256_kernel<<<NROWS / TM, 256, 0, stream>>>(NROWS, x, wtsX, bx, xk1);

    // 4) h_k1 (mode 1): mask, x1acc init, x0 reduce
    fused_ln_mlp_kernel<<<NROWS / TM, 256, 0, stream>>>(
        1, NROWS, xk1, nullptr, nullptr, nullptr, nullptr, nullptr,
        ln1g, ln1b, pe1 + 1 * 256,
        wts1a, m1b + 0, wts1b, m1b + 256, wts1c, m1b + 512,
        x1acc, x0acc, 1.0f, 1.0f, n1, n2, nullptr, nullptr);

    // 5) edge counts
    edge_count_kernel<<<E_EDGES / 256, 256, 0, stream>>>(E_EDGES, w2i, w2i + E_EDGES, c0k2, c1k2);
    edge_count_kernel<<<E_EDGES / 256, 256, 0, stream>>>(E_EDGES, w3i, w3i + E_EDGES, c0k3, c1k3);

    // 6) h_k2 (mode 2): scatter with weights 1/2 (x0) and 1 (x1)
    fused_ln_mlp_kernel<<<E_EDGES / TM, 256, 0, stream>>>(
        2, E_EDGES, nullptr, w2v, W2w, b2w, w2i, w2i + E_EDGES,
        ln1g, ln1b, pe1 + 2 * 256,
        wts1a, m1b + 0, wts1b, m1b + 256, wts1c, m1b + 512,
        x1acc, x0acc, 0.5f, 1.0f, nullptr, nullptr, nullptr, nullptr);

    // 7) h_k3 (mode 2): weights 1/3 (x0) and 1/2 (x1)
    fused_ln_mlp_kernel<<<E_EDGES / TM, 256, 0, stream>>>(
        2, E_EDGES, nullptr, w3v, W3w, b3w, w3i, w3i + E_EDGES,
        ln1g, ln1b, pe1 + 3 * 256,
        wts1a, m1b + 0, wts1b, m1b + 256, wts1c, m1b + 512,
        x1acc, x0acc, 1.0f / 3.0f, 0.5f, nullptr, nullptr, nullptr, nullptr);

    // 8) normalize x0, x1
    x0_fin_kernel<<<16, 256, 0, stream>>>(x0acc, c0k2, c0k3, n1, n2);
    x1_fin_kernel<<<NROWS, 256, 0, stream>>>(x1acc, c1k2, c1k3);

    // 9) x0 += mlp2(ln(x0)+pe2[0])   (in-place, 16 rows)
    fused_ln_mlp_kernel<<<1, 256, 0, stream>>>(
        0, BQ, x0acc, nullptr, nullptr, nullptr, nullptr, nullptr,
        ln2g, ln2b, pe2 + 0 * 256,
        wts2a, m2b + 0, wts2b, m2b + 256, wts2c, m2b + 512,
        x0acc, nullptr, 0.0f, 0.0f, nullptr, nullptr, nullptr, nullptr);

    // 10) x1 += mlp2(ln(x1)+pe2[1])  (in-place)
    fused_ln_mlp_kernel<<<NROWS / TM, 256, 0, stream>>>(
        0, NROWS, x1acc, nullptr, nullptr, nullptr, nullptr, nullptr,
        ln2g, ln2b, pe2 + 1 * 256,
        wts2a, m2b + 0, wts2b, m2b + 256, wts2c, m2b + 512,
        x1acc, nullptr, 0.0f, 0.0f, nullptr, nullptr, nullptr, nullptr);

    // 11) xx = ((x0+x1) + mlp3(ln(x0+x1))) * m + x_k1
    fused_ln_mlp_kernel<<<NROWS / TM, 256, 0, stream>>>(
        3, NROWS, x1acc, nullptr, nullptr, nullptr, nullptr, nullptr,
        ln3g, ln3b, nullptr,
        wts3a, m3b + 0, wts3b, m3b + 256, wts3c, m3b + 512,
        xx, nullptr, 0.0f, 0.0f, n1, n2, x0acc, xk1);

    // 12) out = xx @ Wo + bo
    gemm256_kernel<<<NROWS / TM, 256, 0, stream>>>(NROWS, xx, wtsO, bo, out);
}
